// ForceLayer_33956011442769
// MI455X (gfx1250) — matrix operand order repaired and checked by reading, not compiled, not run on hardware
//
#include <hip/hip_runtime.h>
#include <hip/hip_bf16.h>
#include <stddef.h>
#include <stdint.h>

// ---------------------------------------------------------------------------
// ForceLayer for MI455X (gfx1250): bf16 WMMA everywhere, flash-style attention
// with 3 direction-weighted accumulators. dirs staged per-block via
// GLOBAL_LOAD_ASYNC_TO_LDS (ASYNCcnt) and shared across the 8 heads.
// ---------------------------------------------------------------------------

typedef __attribute__((ext_vector_type(16))) __bf16 v16bf;
typedef __attribute__((ext_vector_type(8)))  __bf16 v8bf;
typedef __attribute__((ext_vector_type(8)))  float  v8f;

union V16U { v16bf v; v8bf h[2]; };

#define B_   4
#define N_   1024
#define D_   512
#define H_   8
#define DH_  64
#define ND3_ (3 * D_)          // 1536

// A-fragment (16x32 bf16): lane holds row M=lane%16; elements 0..15 at
// K = kbase + (lane>=16?16:0) + i (contiguous, matches B fragments).
__device__ __forceinline__ v16bf load_a_frag(const __bf16* rowptr, int kbase, int half8) {
    V16U u;
    u.h[0] = *(const v8bf*)(rowptr + kbase + 2 * half8);
    u.h[1] = *(const v8bf*)(rowptr + kbase + 2 * half8 + 8);
    return u.v;
}

__device__ __forceinline__ v8f wmma_bf16(v16bf a, v16bf b, v8f c) {
    return __builtin_amdgcn_wmma_f32_16x16x32_bf16(false, a, false, b, (short)0, c,
                                                   false, false);
}

// ------------------------------ conversion ---------------------------------
__global__ void conv_emb_kernel(const float* __restrict__ in,
                                unsigned short* __restrict__ outw, int nelem) {
    __bf16* out = (__bf16*)outw;
    int i = blockIdx.x * blockDim.x + threadIdx.x;
    if (i < nelem) out[i] = (__bf16)in[i];
}

// W_proj (D_ x 3D_) row-major  ->  WT (3D_ x D_) bf16 (one W column per row)
__global__ void conv_w_kernel(const float* __restrict__ W,
                              unsigned short* __restrict__ WTw) {
    __bf16* WT = (__bf16*)WTw;
    int i = blockIdx.x * blockDim.x + threadIdx.x;       // i = col*512 + row
    if (i < D_ * ND3_) {
        int row = i & (D_ - 1);
        int col = i >> 9;
        WT[i] = (__bf16)W[(size_t)row * ND3_ + col];
    }
}

// ------------------------------ QKV GEMM -----------------------------------
// (B_*N_ x D_) @ (D_ x 3D_). One wave owns a 32x64 tile: 2 A-frags x 4 B-frags
// -> 8 WMMAs per K-step, ~2.6x less L2 traffic than 16x16 tiles.
__global__ __launch_bounds__(256) void qkv_gemm_kernel(
        const unsigned short* __restrict__ Aw, const unsigned short* __restrict__ WTw,
        const float* __restrict__ bias,
        unsigned short* __restrict__ qw, unsigned short* __restrict__ kw,
        unsigned short* __restrict__ vTw) {
    const __bf16* A  = (const __bf16*)Aw;
    const __bf16* WT = (const __bf16*)WTw;
    __bf16* q  = (__bf16*)qw;
    __bf16* k  = (__bf16*)kw;
    __bf16* vT = (__bf16*)vTw;

    const int lane = threadIdx.x & 31;
    const int wave = threadIdx.x >> 5;
    const int task = blockIdx.x * 8 + wave;              // 3072 tasks
    const int NG   = ND3_ / 64;                           // 24 N-groups
    const int mg   = task / NG;                           // 0..127 (32 rows each)
    const int ng   = task % NG;                           // 0..23  (64 cols each)
    const int half8  = (lane >= 16) ? 8 : 0;
    const int half16 = (lane >= 16) ? 16 : 0;
    const int ln = lane & 15;

    const __bf16* arow0 = A  + (size_t)(mg * 32 + ln) * D_;
    const __bf16* arow1 = arow0 + 16 * D_;
    const __bf16* bbase = WT + (size_t)(ng * 64 + ln) * D_;

    v8f c[8];
#pragma unroll
    for (int t = 0; t < 8; ++t) { v8f z = {}; c[t] = z; }

#pragma unroll 2
    for (int kb = 0; kb < D_; kb += 32) {
        v16bf af0 = load_a_frag(arow0, kb, half8);
        v16bf af1 = load_a_frag(arow1, kb, half8);
#pragma unroll
        for (int t = 0; t < 4; ++t) {
            v16bf bf = *(const v16bf*)(bbase + (size_t)t * 16 * D_ + kb + half16);
            c[t]     = wmma_bf16(af0, bf, c[t]);
            c[4 + t] = wmma_bf16(af1, bf, c[4 + t]);
        }
    }

    // 64-wide N-group stays inside one section and one head (wave-uniform).
    const int sec  = ng / 8;             // 0=q 1=k 2=v
    const int head = ng % 8;
    const float qscale = 0.04419417382415922f;   // 1/sqrt(512)
    float bia[4];
#pragma unroll
    for (int t = 0; t < 4; ++t) bia[t] = bias[ng * 64 + t * 16 + ln];

#pragma unroll
    for (int part = 0; part < 2; ++part)
#pragma unroll
        for (int t = 0; t < 4; ++t) {
            const int dcol = t * 16 + ln;
#pragma unroll
            for (int r = 0; r < 8; ++r) {
                int grow = mg * 32 + part * 16 + r + half8;
                int bidx = grow >> 10;
                int i    = grow & (N_ - 1);
                float val = c[part * 4 + t][r] + bia[t];
                size_t bh = (size_t)(bidx * H_ + head);
                if (sec == 0)      q [(bh * N_ + i) * DH_ + dcol] = (__bf16)(val * qscale);
                else if (sec == 1) k [(bh * N_ + i) * DH_ + dcol] = (__bf16)val;
                else               vT[(bh * DH_ + dcol) * N_ + i] = (__bf16)val;
            }
        }
}

// ------------------------------ attention ----------------------------------
// Block = one (b, i-tile), 8 waves = 8 heads. dirs tile async-staged to LDS
// once per key-block and shared by all heads. Online softmax per wave.
__global__ __launch_bounds__(256) void attn_kernel(
        const unsigned short* __restrict__ qw, const unsigned short* __restrict__ kw,
        const unsigned short* __restrict__ vTw,
        const float* __restrict__ d_mask, const float* __restrict__ dirs,
        const float* __restrict__ w_force, float* __restrict__ partial) {
    const __bf16* qg  = (const __bf16*)qw;
    const __bf16* kg  = (const __bf16*)kw;
    const __bf16* vTg = (const __bf16*)vTw;

    __shared__ float  lds_dirs[16 * 32 * 3];       // 6 KB: 16 rows x 32 cols x xyz
    __shared__ __bf16 lds_p[8][16 * 32];           // 8 KB: per-wave P staging

    const int lane  = threadIdx.x & 31;
    const int wave  = threadIdx.x >> 5;            // = head
    const int bidx  = blockIdx.x >> 6;             // batch
    const int itile = blockIdx.x & 63;
    const int h     = wave;
    const int bh    = bidx * H_ + h;
    const int half8  = (lane >= 16) ? 8 : 0;
    const int half16 = (lane >= 16) ? 16 : 0;
    const int ln     = lane & 15;
    const int i0     = itile * 16;

    const __bf16* qbh = qg  + (size_t)bh * N_ * DH_;
    const __bf16* kbh = kg  + (size_t)bh * N_ * DH_;
    const __bf16* vbh = vTg + (size_t)bh * DH_ * N_;
    const float*  mbh = d_mask + (size_t)bh * N_ * N_;
    const uint64_t dirb = (uint64_t)(uintptr_t)(dirs + (size_t)bidx * N_ * N_ * 3);
    __bf16* myp = lds_p[wave];
    const unsigned ldsd_base = (unsigned)(uintptr_t)(&lds_dirs[0]);

    // q A-fragments for this 16-row tile (K = dh = 64 -> 2 fragments)
    const __bf16* qrow = qbh + (size_t)(i0 + ln) * DH_;
    v16bf aq0 = load_a_frag(qrow, 0,  half8);
    v16bf aq1 = load_a_frag(qrow, 32, half8);

    float m[8], l[8];
    v8f O[12];                                      // [axis*4 + dh_tile]
#pragma unroll
    for (int r = 0; r < 8; ++r) { m[r] = -1e30f; l[r] = 0.0f; }
#pragma unroll
    for (int t = 0; t < 12; ++t) { v8f z = {}; O[t] = z; }

    for (int j0 = 0; j0 < N_; j0 += 32) {
        // ---- all waves done reading previous dirs tile ----
        __syncthreads();

        // ---- cooperative async DMA of dirs[i0:i0+16, j0:j0+32, 0:3] ----
        // 16 rows x 384 B = 768 x 8 B chunks; 8 waves x 3 instr x 32 lanes.
#pragma unroll
        for (int t = 0; t < 3; ++t) {
            int cidx = (wave * 3 + t) * 32 + lane;          // 0..767
            int row  = cidx / 48;                           // 48 x 8B per row
            int off8 = cidx - row * 48;
            unsigned lds_addr = ldsd_base + row * 384 + off8 * 8;
            uint64_t gaddr = dirb + ((size_t)(i0 + row) * N_ + j0) * 12ull
                                  + (unsigned)(off8 * 8);
            asm volatile("global_load_async_to_lds_b64 %0, %1, off"
                         :: "v"(lds_addr), "v"(gaddr) : "memory");
        }

        // ---- S = q @ k^T (+mask), overlapped with the DMA: 4 WMMAs ----
        v8f S[2];
#pragma unroll
        for (int jj = 0; jj < 2; ++jj) {
            const __bf16* kcol = kbh + (size_t)(j0 + jj * 16 + ln) * DH_;
            v16bf kb0 = *(const v16bf*)(kcol + half16);
            v16bf kb1 = *(const v16bf*)(kcol + 32 + half16);
            v8f cc = {};
            cc = wmma_bf16(aq0, kb0, cc);
            cc = wmma_bf16(aq1, kb1, cc);
            const float* mrow = mbh + (size_t)i0 * N_ + (j0 + jj * 16 + ln);
#pragma unroll
            for (int r = 0; r < 8; ++r)
                cc[r] += mrow[(size_t)(r + half8) * N_];
            S[jj] = cc;
        }

        // ---- online softmax update over these 32 columns ----
        float alpha[8];
#pragma unroll
        for (int r = 0; r < 8; ++r) {
            float x = fmaxf(S[0][r], S[1][r]);
#pragma unroll
            for (int s = 8; s >= 1; s >>= 1) x = fmaxf(x, __shfl_xor(x, s, 32));
            float mn = fmaxf(m[r], x);
            alpha[r] = __expf(m[r] - mn);
            float p0 = __expf(S[0][r] - mn);
            float p1 = __expf(S[1][r] - mn);
            S[0][r] = p0; S[1][r] = p1;
            float sum = p0 + p1;
#pragma unroll
            for (int s = 8; s >= 1; s >>= 1) sum += __shfl_xor(sum, s, 32);
            l[r] = l[r] * alpha[r] + sum;
            m[r] = mn;
        }
#pragma unroll
        for (int t = 0; t < 12; ++t)
#pragma unroll
            for (int r = 0; r < 8; ++r) O[t][r] *= alpha[r];

        // ---- V B-fragments (K = j0..j0+31), reused for 3 axes ----
        v16bf vb[4];
#pragma unroll
        for (int t = 0; t < 4; ++t)
            vb[t] = *(const v16bf*)(vbh + (size_t)(t * 16 + ln) * N_ + j0 + half16);

        // ---- dirs tile ready ----
        asm volatile("s_wait_asynccnt 0" ::: "memory");
        __syncthreads();

        // ---- per axis: P_a = P * dirs_a -> bf16 A-frag via LDS -> 4 WMMAs ----
#pragma unroll
        for (int a = 0; a < 3; ++a) {
#pragma unroll
            for (int jj = 0; jj < 2; ++jj)
#pragma unroll
                for (int r = 0; r < 8; ++r) {
                    float d = lds_dirs[(r + half8) * 96 + (jj * 16 + ln) * 3 + a];
                    myp[(r + half8) * 32 + jj * 16 + ln] = (__bf16)(S[jj][r] * d);
                }
            asm volatile("s_wait_dscnt 0" ::: "memory");   // cross-lane LDS w->r
            V16U pa;
            pa.h[0] = *(const v8bf*)(myp + ln * 32 + half16);
            pa.h[1] = *(const v8bf*)(myp + ln * 32 + half16 + 8);
#pragma unroll
            for (int t = 0; t < 4; ++t)
                O[a * 4 + t] = wmma_bf16(pa.v, vb[t], O[a * 4 + t]);
            asm volatile("s_wait_dscnt 0" ::: "memory");   // reads retire before reuse
        }
    }

    // ---- finalize: O /= l, dot with w_force chunk, per-head partial ----
    float wf[4];
#pragma unroll
    for (int t = 0; t < 4; ++t) wf[t] = w_force[h * DH_ + t * 16 + ln];
#pragma unroll
    for (int a = 0; a < 3; ++a)
#pragma unroll
        for (int r = 0; r < 8; ++r) {
            float acc = 0.0f;
#pragma unroll
            for (int t = 0; t < 4; ++t) acc += O[a * 4 + t][r] * wf[t];
            acc *= (1.0f / l[r]);
#pragma unroll
            for (int s = 8; s >= 1; s >>= 1) acc += __shfl_xor(acc, s, 32);
            if (ln == 0)
                partial[((size_t)bh * N_ + i0 + r + half8) * 3 + a] = acc;
        }
}

// ------------------------- deterministic head reduce -----------------------
__global__ void reduce_kernel(const float* __restrict__ partial,
                              const float* __restrict__ b_force,
                              float* __restrict__ out) {
    int idx = blockIdx.x * blockDim.x + threadIdx.x;     // (b, i, a) flat
    if (idx >= B_ * N_ * 3) return;
    int a = idx % 3;
    int t = idx / 3;
    int i = t & (N_ - 1);
    int b = t >> 10;
    float acc = b_force[0];
#pragma unroll
    for (int h = 0; h < H_; ++h)
        acc += partial[(((size_t)(b * H_ + h)) * N_ + i) * 3 + a];
    out[idx] = acc;
}

// ------------------------------- launcher ----------------------------------
extern "C" void kernel_launch(void* const* d_in, const int* in_sizes, int n_in,
                              void* d_out, int out_size, void* d_ws, size_t ws_size,
                              hipStream_t stream) {
    const float* emb     = (const float*)d_in[0];
    const float* dirs    = (const float*)d_in[1];
    const float* d_mask  = (const float*)d_in[2];
    const float* W_proj  = (const float*)d_in[3];
    const float* b_proj  = (const float*)d_in[4];
    const float* w_force = (const float*)d_in[5];
    const float* b_force = (const float*)d_in[6];

    char* ws = (char*)d_ws;
    const size_t SZ_EMB = (size_t)B_ * N_ * D_ * 2;       // 4 MB bf16
    const size_t SZ_WT  = (size_t)D_ * ND3_ * 2;          // 1.5 MB bf16
    const size_t SZ_QKV = (size_t)B_ * H_ * N_ * DH_ * 2; // 4 MB each
    unsigned short* embb = (unsigned short*)(ws);
    unsigned short* WTb  = (unsigned short*)(ws + SZ_EMB);
    unsigned short* qb   = (unsigned short*)(ws + SZ_EMB + SZ_WT);
    unsigned short* kb   = (unsigned short*)(ws + SZ_EMB + SZ_WT + SZ_QKV);
    unsigned short* vTb  = (unsigned short*)(ws + SZ_EMB + SZ_WT + 2 * SZ_QKV);
    float* partial       = (float*)(ws + SZ_EMB + SZ_WT + 3 * SZ_QKV);

    int n_emb = B_ * N_ * D_;                // 2,097,152
    int n_w   = D_ * ND3_;                   //   786,432
    conv_emb_kernel<<<(n_emb + 255) / 256, 256, 0, stream>>>(emb, embb, n_emb);
    conv_w_kernel  <<<(n_w   + 255) / 256, 256, 0, stream>>>(W_proj, WTb);

    // 128 M-groups (32 rows) x 24 N-groups (64 cols) = 3072 waves, 8/block
    qkv_gemm_kernel<<<384, 256, 0, stream>>>(embb, WTb, b_proj, qb, kb, vTb);

    // 4 batches x 64 i-tiles = 256 blocks; 8 waves/block = 8 heads
    attn_kernel<<<256, 256, 0, stream>>>(qb, kb, vTb, d_mask, dirs, w_force, partial);

    reduce_kernel<<<(B_ * N_ * 3 + 255) / 256, 256, 0, stream>>>(partial, b_force,
                                                                 (float*)d_out);
}
